// TGNMemory_82944408420703
// MI455X (gfx1250) — matrix operand order repaired and checked by reading, not compiled, not run on hardware
//
#include <hip/hip_runtime.h>
#include <cstdint>
#include <cstddef>

typedef __attribute__((ext_vector_type(16))) __bf16 v16bf;
typedef __attribute__((ext_vector_type(8)))  __bf16 v8bf;
typedef __attribute__((ext_vector_type(8)))  float  v8f;

#define NUM_NODES 200000
#define NODE_DIM  256
#define MSG_DIM   685
#define H3        768
#define NUM_UPD   65536
#define KPAD      704   /* 22 * 32, zero-padded K for W_ih / messages */
#define TILE_M    32    /* updates per workgroup (two 16-row WMMA tiles) */

__device__ __forceinline__ __bf16 f2bf(float f) {
  // round-to-nearest-even f32 -> bf16
  uint32_t u = __builtin_bit_cast(uint32_t, f);
  uint32_t r = (u + 0x7FFFu + ((u >> 16) & 1u)) >> 16;
  uint16_t h = (uint16_t)r;
  return __builtin_bit_cast(__bf16, h);
}

__device__ __forceinline__ float sigmoid_f(float x) {
  return 1.0f / (1.0f + __expf(-x));
}

// Build a 16-element bf16 fragment from two aligned 16-byte groups.
// Used for both A (LDS) and B (global) fragments:
//   A 16x32 layout: lane<16 -> K {kb..kb+15}; lane>=16 -> K {kb+16..kb+31}
//   B 32x16 layout: lane<16 -> K {kb..kb+15}; lane>=16 -> K {kb+16..kb+31}
__device__ __forceinline__ v16bf frag16(const __bf16* p0, const __bf16* p1) {
  v8bf lo = *(const v8bf*)p0;
  v8bf up = *(const v8bf*)p1;
  v16bf f;
  #pragma unroll
  for (int i = 0; i < 8; ++i) { f[i] = lo[i]; f[i + 8] = up[i]; }
  return f;
}

#define WMMA_BF16(ACC, A, B) \
  (ACC) = __builtin_amdgcn_wmma_f32_16x16x32_bf16(false, (A), false, (B), (short)0, (ACC), false, false)

// ---- Prep: weights f32 -> bf16, W_ih K-padded 685 -> 704 (rows 16B aligned) ----
__global__ __launch_bounds__(256)
void prep_weights_kernel(const float* __restrict__ W_ih,
                         const float* __restrict__ W_hh,
                         __bf16* __restrict__ wih,   // [768][704]
                         __bf16* __restrict__ whh) { // [768][256]
  const int stride = (int)(gridDim.x * blockDim.x);
  const int g = (int)(blockIdx.x * blockDim.x + threadIdx.x);
  for (int i = g; i < H3 * KPAD; i += stride) {
    const int r = i / KPAD, c = i - r * KPAD;
    wih[i] = (c < MSG_DIM) ? f2bf(W_ih[(size_t)r * MSG_DIM + c]) : f2bf(0.0f);
  }
  for (int i = g; i < H3 * NODE_DIM; i += stride) {
    whh[i] = f2bf(W_hh[i]);
  }
}

// ---- Main fused GRU update ----
// grid = (NUM_UPD/32, 2); block = 256 (8 waves).
// Each wave owns one 16-column tile per gate: gate-relative cols
// [128*blockIdx.y + 16*wave, +16). Two M sub-tiles share every B fragment.
__global__ __launch_bounds__(256)
void tgn_gru_wmma_kernel(const float* __restrict__ memory,
                         const float* __restrict__ messages,
                         const int*   __restrict__ node_idx,
                         const __bf16* __restrict__ wih,
                         const __bf16* __restrict__ whh,
                         const float* __restrict__ b_ih,
                         const float* __restrict__ b_hh,
                         float* __restrict__ out) {
  __shared__ __bf16 sMsg[TILE_M * KPAD];      // 45056 B
  __shared__ __bf16 sH16[TILE_M * NODE_DIM];  // 16384 B
  __shared__ int    sIdx[TILE_M];

  const int tid  = (int)threadIdx.x;
  const int wave = tid >> 5;
  const int lane = tid & 31;
  const int n0   = lane & 15;
  const int hi   = lane >> 4;
  const int u0   = (int)blockIdx.x * TILE_M;

  if (tid < TILE_M) sIdx[tid] = node_idx[u0 + tid];

  // Stage message tile: f32 -> bf16, zero-pad K to 704.
  for (int i = tid; i < TILE_M * KPAD; i += 256) {
    const int r = i / KPAD, cc = i - r * KPAD;
    const float v = (cc < MSG_DIM) ? messages[(size_t)(u0 + r) * MSG_DIM + cc] : 0.0f;
    sMsg[i] = f2bf(v);
  }
  __syncthreads();  // sIdx + sMsg ready

  // Stage gathered h tile (bf16 for the GEMM).
  for (int i = tid; i < TILE_M * NODE_DIM; i += 256) {
    const int r = i >> 8, cc = i & (NODE_DIM - 1);
    sH16[i] = f2bf(memory[(size_t)sIdx[r] * NODE_DIM + cc]);
  }
  __syncthreads();

  // This lane's column within each gate third.
  const int c = (int)blockIdx.y * 128 + wave * 16 + n0;   // [0,256)

  const __bf16* Wi_r = wih + (size_t)(c)       * KPAD;
  const __bf16* Wi_z = wih + (size_t)(256 + c) * KPAD;
  const __bf16* Wi_n = wih + (size_t)(512 + c) * KPAD;
  const __bf16* Wh_r = whh + (size_t)(c)       * NODE_DIM;
  const __bf16* Wh_z = whh + (size_t)(256 + c) * NODE_DIM;
  const __bf16* Wh_n = whh + (size_t)(512 + c) * NODE_DIM;

  // r/z fuse gi+gh into one accumulator; n keeps i_n / h_n separate.
  v8f ar0 = {}, ar1 = {}, az0 = {}, az1 = {};
  v8f ani0 = {}, ani1 = {}, anh0 = {}, anh1 = {};

  const __bf16* a0row = sMsg + n0 * KPAD;          // M rows 0..15
  const __bf16* a1row = sMsg + (16 + n0) * KPAD;   // M rows 16..31

  // ---- GEMM 1: gi = messages @ W_ih^T (22 branch-free K chunks) ----
  for (int kc = 0; kc < KPAD / 32; ++kc) {
    const int kb = kc * 32;
    const int ka = kb + hi * 16;
    const int kbb = kb + hi * 16;
    const v16bf a0 = frag16(a0row + ka, a0row + ka + 8);
    const v16bf a1 = frag16(a1row + ka, a1row + ka + 8);
    v16bf b;
    b = frag16(Wi_r + kbb, Wi_r + kbb + 8);
    WMMA_BF16(ar0, a0, b);  WMMA_BF16(ar1, a1, b);
    b = frag16(Wi_z + kbb, Wi_z + kbb + 8);
    WMMA_BF16(az0, a0, b);  WMMA_BF16(az1, a1, b);
    b = frag16(Wi_n + kbb, Wi_n + kbb + 8);
    WMMA_BF16(ani0, a0, b); WMMA_BF16(ani1, a1, b);
  }

  // ---- GEMM 2: gh = h @ W_hh^T (8 K chunks) ----
  const __bf16* h0row = sH16 + n0 * NODE_DIM;
  const __bf16* h1row = sH16 + (16 + n0) * NODE_DIM;
  for (int kc = 0; kc < NODE_DIM / 32; ++kc) {
    const int kb = kc * 32;
    const int ka = kb + hi * 16;
    const int kbb = kb + hi * 16;
    const v16bf a0 = frag16(h0row + ka, h0row + ka + 8);
    const v16bf a1 = frag16(h1row + ka, h1row + ka + 8);
    v16bf b;
    b = frag16(Wh_r + kbb, Wh_r + kbb + 8);
    WMMA_BF16(ar0, a0, b);  WMMA_BF16(ar1, a1, b);
    b = frag16(Wh_z + kbb, Wh_z + kbb + 8);
    WMMA_BF16(az0, a0, b);  WMMA_BF16(az1, a1, b);
    b = frag16(Wh_n + kbb, Wh_n + kbb + 8);
    WMMA_BF16(anh0, a0, b); WMMA_BF16(anh1, a1, b);
  }

  // ---- Gate math + scatter. C/D layout: VGPR v -> M = v + 8*hi, col = n0. ----
  const float bir = b_ih[c],       bhr = b_hh[c];
  const float biz = b_ih[256 + c], bhz = b_hh[256 + c];
  const float bin = b_ih[512 + c], bhn = b_hh[512 + c];

  #pragma unroll
  for (int mt = 0; mt < 2; ++mt) {
    const v8f vr  = mt ? ar1  : ar0;
    const v8f vz  = mt ? az1  : az0;
    const v8f vni = mt ? ani1 : ani0;
    const v8f vnh = mt ? anh1 : anh0;
    #pragma unroll
    for (int v = 0; v < 8; ++v) {
      const int    row  = mt * 16 + v + hi * 8;
      const size_t node = (size_t)sIdx[row];
      const float  rg   = sigmoid_f(vr[v] + bir + bhr);
      const float  zg   = sigmoid_f(vz[v] + biz + bhz);
      const float  ng   = tanhf(vni[v] + bin + rg * (vnh[v] + bhn));
      const float  hp   = memory[node * NODE_DIM + c];   // L2/L1-hot (just staged)
      out[node * NODE_DIM + c] = (1.0f - zg) * ng + zg * hp;
    }
  }
}

extern "C" void kernel_launch(void* const* d_in, const int* in_sizes, int n_in,
                              void* d_out, int out_size, void* d_ws, size_t ws_size,
                              hipStream_t stream) {
  const float* memory   = (const float*)d_in[0];
  const float* messages = (const float*)d_in[1];
  const int*   node_idx = (const int*)  d_in[2];
  const float* W_ih     = (const float*)d_in[3];
  const float* W_hh     = (const float*)d_in[4];
  const float* b_ih     = (const float*)d_in[5];
  const float* b_hh     = (const float*)d_in[6];
  float*       out      = (float*)d_out;

  __bf16* wih = (__bf16*)d_ws;                 // [768][704] bf16 = 1,081,344 B
  __bf16* whh = wih + (size_t)H3 * KPAD;       // [768][256] bf16 =   393,216 B

  // 1) Base copy: out = memory (graph-capture-safe async d2d).
  hipMemcpyAsync(out, memory, (size_t)NUM_NODES * NODE_DIM * sizeof(float),
                 hipMemcpyDeviceToDevice, stream);

  // 2) Weight conversion/padding (tiny; keeps the GEMM loops branch-free).
  prep_weights_kernel<<<dim3(512), dim3(256), 0, stream>>>(W_ih, W_hh, wih, whh);

  // 3) Fused GRU update: 2048 M-tiles x 2 column halves.
  dim3 grid(NUM_UPD / TILE_M, 2);
  dim3 block(256);
  tgn_gru_wmma_kernel<<<grid, block, 0, stream>>>(
      memory, messages, node_idx, wih, whh, b_ih, b_hh, out);
}
